// LlamaAttention_40475771797770
// MI455X (gfx1250) — compile-verified, matrix operand order checked by reading, not run on hardware
//
#include <hip/hip_runtime.h>
#include <hip/hip_bf16.h>
#include <math.h>
#include <stddef.h>

typedef __attribute__((ext_vector_type(16))) _Float16 v16h;
typedef __attribute__((ext_vector_type(8)))  _Float16 v8h;
typedef __attribute__((ext_vector_type(4)))  _Float16 v4h;
typedef __attribute__((ext_vector_type(8)))  float    v8f;
typedef __attribute__((ext_vector_type(4)))  int      v4i;

#define S_LEN 2048
#define HIDN  2048
#define NHEAD 32
#define NKV   8
#define DHEAD 64

#ifndef __has_builtin
#define __has_builtin(x) 0
#endif

#if __has_builtin(__builtin_amdgcn_global_load_async_to_lds_b128)
#define HAVE_ASYNC_LDS 1
#else
#define HAVE_ASYNC_LDS 0
#endif

typedef __attribute__((address_space(1))) v4i* gv4p;   // global b128
typedef __attribute__((address_space(3))) v4i* lv4p;   // LDS b128

// 16B global -> LDS copy; async (ASYNCcnt-tracked, no VGPR round trip) when available
__device__ inline void cp16_g2l(const _Float16* g, _Float16* l) {
#if HAVE_ASYNC_LDS
  __builtin_amdgcn_global_load_async_to_lds_b128((gv4p)(v4i*)g, (lv4p)(v4i*)l, 0, 0);
#else
  *(v8h*)l = *(const v8h*)g;
#endif
}
__device__ inline void async_wait0() {
#if __has_builtin(__builtin_amdgcn_s_wait_asynccnt)
  __builtin_amdgcn_s_wait_asynccnt(0);
#elif HAVE_ASYNC_LDS
  asm volatile("s_wait_asynccnt 0x0" ::: "memory");
#endif
}

// ---------------------------------------------------------------------------
// WMMA helper (gfx1250: wave32, 16x16x32 f16 -> f32)
// ---------------------------------------------------------------------------
__device__ inline v8f wmma_f16(v16h a, v16h b, v8f c) {
  return __builtin_amdgcn_wmma_f32_16x16x32_f16(
      /*neg_a=*/false, a, /*neg_b=*/false, b,
      /*c_mod=*/(short)0, c, /*reuse_a=*/false, /*reuse_b=*/false);
}

// Load a 16x32 f16 fragment (A- or B-operand layout) from a K-major matrix.
__device__ inline v16h frag_ld(const _Float16* __restrict__ base, int ld) {
  const int lane = threadIdx.x & 31;
  const int r    = lane & 15;
  const int kh   = (lane >> 4) << 3;
  const _Float16* p = base + r * ld + kh;
  v8h a0 = *(const v8h*)(p);
  v8h a1 = *(const v8h*)(p + 16);
  return __builtin_shufflevector(a0, a1, 0,1,2,3,4,5,6,7,8,9,10,11,12,13,14,15);
}

// ---------------------------------------------------------------------------
// DPP row_ror butterfly reductions over each 16-lane row (no LDS traffic)
// ---------------------------------------------------------------------------
template<int CTRL>
__device__ inline float dpp_rotf(float x) {
  int xi = __builtin_bit_cast(int, x);
  int r  = __builtin_amdgcn_update_dpp(xi, xi, CTRL, 0xf, 0xf, true);
  return __builtin_bit_cast(float, r);
}
__device__ inline float row_max16(float x) {
  x = fmaxf(x, dpp_rotf<0x121>(x));   // row_ror:1
  x = fmaxf(x, dpp_rotf<0x122>(x));   // row_ror:2
  x = fmaxf(x, dpp_rotf<0x124>(x));   // row_ror:4
  x = fmaxf(x, dpp_rotf<0x128>(x));   // row_ror:8
  return x;
}
__device__ inline float row_sum16(float x) {
  x += dpp_rotf<0x121>(x);
  x += dpp_rotf<0x122>(x);
  x += dpp_rotf<0x124>(x);
  x += dpp_rotf<0x128>(x);
  return x;
}

// ---------------------------------------------------------------------------
// fp32 -> f16 conversion
// ---------------------------------------------------------------------------
__global__ __launch_bounds__(256) void cvt_kernel(const float* __restrict__ src,
                                                  _Float16* __restrict__ dst, int n) {
  int i = (blockIdx.x * blockDim.x + threadIdx.x) * 4;
  if (i + 3 < n) {
    float4 v = *(const float4*)(src + i);
    v4h o = {(_Float16)v.x, (_Float16)v.y, (_Float16)v.z, (_Float16)v.w};
    *(v4h*)(dst + i) = o;
  } else {
    for (; i < n; ++i) dst[i] = (_Float16)src[i];
  }
}

// ---------------------------------------------------------------------------
// GEMM: C[M,N] = A[M,K] * B[N,K]^T   (f16 K-major in; f16 or f32 out)
// block tile 128x128, 256 threads (8 waves, 2x4), wave tile 64x32,
// K-tile 64, double-buffered LDS, async global->LDS staging.
// ---------------------------------------------------------------------------
template<bool OUT_F32>
__global__ __launch_bounds__(256) void gemm_nt(
    const _Float16* __restrict__ A, const _Float16* __restrict__ B,
    void* __restrict__ Cp, int M, int N, int K, int lda, int ldb, int ldc) {
  __shared__ _Float16 sA[2][128][72];
  __shared__ _Float16 sB[2][128][72];
  const int t    = threadIdx.x;
  const int lane = t & 31;
  const int wid  = t >> 5;
  const int wm   = wid >> 2;          // 0..1
  const int wn   = wid & 3;           // 0..3
  const int bm   = blockIdx.y * 128;
  const int bn   = blockIdx.x * 128;

  v8f acc[4][2];
#pragma unroll
  for (int i = 0; i < 4; ++i)
#pragma unroll
    for (int j = 0; j < 2; ++j) acc[i][j] = (v8f){};

  const int lrow = t >> 1;            // 0..127
  const int lcol = (t & 1) << 5;      // 0 or 32

  auto stage = [&](int buf, int kb) {
    const _Float16* ga = A + (size_t)(bm + lrow) * lda + kb + lcol;
    const _Float16* gb = B + (size_t)(bn + lrow) * ldb + kb + lcol;
    _Float16* la = &sA[buf][lrow][lcol];
    _Float16* lb = &sB[buf][lrow][lcol];
#pragma unroll
    for (int c = 0; c < 4; ++c) {
      cp16_g2l(ga + 8 * c, la + 8 * c);
      cp16_g2l(gb + 8 * c, lb + 8 * c);
    }
  };

  stage(0, 0);
  async_wait0();
  __syncthreads();

  int buf = 0;
  for (int kb = 0; kb < K; kb += 64) {
    if (kb + 64 < K) stage(buf ^ 1, kb + 64);   // prefetch next tile (async)
#pragma unroll
    for (int kk = 0; kk < 2; ++kk) {
      v16h bf0 = frag_ld(&sB[buf][wn * 32][kk * 32],      72);
      v16h bf1 = frag_ld(&sB[buf][wn * 32 + 16][kk * 32], 72);
#pragma unroll
      for (int i = 0; i < 4; ++i) {
        v16h af = frag_ld(&sA[buf][wm * 64 + i * 16][kk * 32], 72);
        acc[i][0] = wmma_f16(af, bf0, acc[i][0]);
        acc[i][1] = wmma_f16(af, bf1, acc[i][1]);
      }
    }
    async_wait0();
    __syncthreads();
    buf ^= 1;
  }

  const int half = lane >> 4;
  const int cl   = lane & 15;
#pragma unroll
  for (int i = 0; i < 4; ++i)
#pragma unroll
    for (int j = 0; j < 2; ++j) {
      const int mbase = bm + wm * 64 + i * 16 + half * 8;
      const int nn    = bn + wn * 32 + j * 16 + cl;
#pragma unroll
      for (int r = 0; r < 8; ++r) {
        float v = acc[i][j][r];
        if (OUT_F32) ((float*)Cp)[(size_t)(mbase + r) * ldc + nn] = v;
        else ((_Float16*)Cp)[(size_t)(mbase + r) * ldc + nn] = (_Float16)v;
      }
    }
}

// ---------------------------------------------------------------------------
// RoPE applied in-place to Q (S x H*D) and K (S x KV*D), f16
// ---------------------------------------------------------------------------
__global__ __launch_bounds__(256) void rope_kernel(
    _Float16* __restrict__ Q, _Float16* __restrict__ Kb,
    const float* __restrict__ powers, const int* __restrict__ pos_ids) {
  const int PAIRS = (NHEAD + NKV) * (DHEAD / 2);   // 40 * 32
  int idx = blockIdx.x * blockDim.x + threadIdx.x;
  if (idx >= S_LEN * PAIRS) return;
  int s    = idx / PAIRS;
  int rem  = idx - s * PAIRS;
  int slot = rem >> 5;       // 0..39
  int d    = rem & 31;       // 0..31

  float pos  = (float)pos_ids[s];
  float w    = powers[d];
  float sg   = 1.0f / (1.0f + __expf(-w));
  float invf = __expf(-sg * 9.210340371976184f);   // ln(10000)
  float fr   = pos * invf;
  float c    = __cosf(fr);
  float sn   = __sinf(fr);

  _Float16* base;
  if (slot < NHEAD) base = Q  + (size_t)s * HIDN          + slot * DHEAD;
  else              base = Kb + (size_t)s * (NKV * DHEAD) + (slot - NHEAD) * DHEAD;

  float x1 = (float)base[d];
  float x2 = (float)base[d + 32];
  base[d]      = (_Float16)(x1 * c - x2 * sn);
  base[d + 32] = (_Float16)(x2 * c + x1 * sn);
}

// ---------------------------------------------------------------------------
// Fused causal flash-attention.
// grid = (S/128, H); 256 threads = 8 waves; wave w -> q rows qb*128+w*16 .. +15
// ---------------------------------------------------------------------------
__global__ __launch_bounds__(256) void attn_fused(
    const _Float16* __restrict__ Q, const _Float16* __restrict__ Kb,
    const _Float16* __restrict__ Vb, _Float16* __restrict__ O) {
  __shared__ _Float16 sK[32][72];        // 32 keys x 64 d  (padded)
  __shared__ _Float16 sVt[64][40];       // transposed: 64 d x 32 keys (padded)
  __shared__ _Float16 sP[8][16][40];     // per-wave P scratch 16 x 32 (padded)

  const int t    = threadIdx.x;
  const int lane = t & 31;
  const int wid  = t >> 5;
  const int qb   = blockIdx.x;
  const int h    = blockIdx.y;
  const int kvh  = h >> 2;               // N_REP = 4
  const int q0   = qb * 128 + wid * 16;
  const int half = lane >> 4;
  const int cl   = lane & 15;

  const _Float16* qp = Q + (size_t)q0 * HIDN + h * DHEAD;
  v16h aq0 = frag_ld(qp, HIDN);
  v16h aq1 = frag_ld(qp + 32, HIDN);

  v8f o0 = (v8f){}, o1 = (v8f){}, o2 = (v8f){}, o3 = (v8f){};
  float mi[8], li[8];
#pragma unroll
  for (int r = 0; r < 8; ++r) { mi[r] = -3.0e38f; li[r] = 0.0f; }

  const int ntile = (qb + 1) * 4;        // causal upper bound for the block
  for (int ti = 0; ti < ntile; ++ti) {
    const int kb = ti * 32;
    {
      const int row = t >> 3;            // 0..31 (key)
      const int ch  = (t & 7) << 3;      // 0,8,..,56 (d)
      // K tile: async straight into LDS
      cp16_g2l(Kb + (size_t)(kb + row) * (NKV * DHEAD) + kvh * DHEAD + ch,
               &sK[row][ch]);
      // V tile: transposed on the way in (needs the VGPR hop)
      v8h vv = *(const v8h*)(Vb + (size_t)(kb + row) * (NKV * DHEAD) + kvh * DHEAD + ch);
#pragma unroll
      for (int e = 0; e < 8; ++e) sVt[ch + e][row] = vv[e];
    }
    async_wait0();
    __syncthreads();

    if (kb <= q0 + 15) {                 // wave-uniform predicate
      v8f s0 = (v8f){}, s1 = (v8f){};
      v16h bk00 = frag_ld(&sK[0][0],   72);
      v16h bk01 = frag_ld(&sK[0][32],  72);
      v16h bk10 = frag_ld(&sK[16][0],  72);
      v16h bk11 = frag_ld(&sK[16][32], 72);
      s0 = wmma_f16(aq0, bk00, s0);
      s0 = wmma_f16(aq1, bk01, s0);
      s1 = wmma_f16(aq0, bk10, s1);
      s1 = wmma_f16(aq1, bk11, s1);

      // online softmax (f32); VGPR slot r holds rows r (lanes 0-15) / r+8 (16-31)
#pragma unroll
      for (int r = 0; r < 8; ++r) {
        const int qrow = q0 + r + half * 8;
        float a0 = s0[r] * 0.125f;       // 1/sqrt(64)
        float a1 = s1[r] * 0.125f;
        if (kb + cl      > qrow) a0 = -3.0e38f;
        if (kb + 16 + cl > qrow) a1 = -3.0e38f;
        float mx    = row_max16(fmaxf(a0, a1));
        float mnew  = fmaxf(mi[r], mx);
        float alpha = __expf(mi[r] - mnew);
        float p0 = __expf(a0 - mnew);
        float p1 = __expf(a1 - mnew);
        float rs = row_sum16(p0 + p1);
        li[r] = li[r] * alpha + rs;
        mi[r] = mnew;
        o0[r] *= alpha; o1[r] *= alpha; o2[r] *= alpha; o3[r] *= alpha;
        const int prow = r + half * 8;
        sP[wid][prow][cl]      = (_Float16)p0;
        sP[wid][prow][16 + cl] = (_Float16)p1;
      }

      // O += P(16x32) * V(32x64) ; V consumed transposed from LDS
      v16h pa  = frag_ld(&sP[wid][0][0], 40);
      v16h bv0 = frag_ld(&sVt[0][0],  40);
      v16h bv1 = frag_ld(&sVt[16][0], 40);
      v16h bv2 = frag_ld(&sVt[32][0], 40);
      v16h bv3 = frag_ld(&sVt[48][0], 40);
      o0 = wmma_f16(pa, bv0, o0);
      o1 = wmma_f16(pa, bv1, o1);
      o2 = wmma_f16(pa, bv2, o2);
      o3 = wmma_f16(pa, bv3, o3);
    }
    __syncthreads();
  }

#pragma unroll
  for (int r = 0; r < 8; ++r) {
    const size_t row = (size_t)(q0 + r + half * 8);
    const float invl = 1.0f / li[r];
    _Float16* op = O + row * HIDN + h * DHEAD;
    op[0  + cl] = (_Float16)(o0[r] * invl);
    op[16 + cl] = (_Float16)(o1[r] * invl);
    op[32 + cl] = (_Float16)(o2[r] * invl);
    op[48 + cl] = (_Float16)(o3[r] * invl);
  }
}

// ---------------------------------------------------------------------------
// launcher
// ---------------------------------------------------------------------------
extern "C" void kernel_launch(void* const* d_in, const int* in_sizes, int n_in,
                              void* d_out, int out_size, void* d_ws, size_t ws_size,
                              hipStream_t stream) {
  const float* hs     = (const float*)d_in[0];
  const int*   posids = (const int*)d_in[1];
  const float* powers = (const float*)d_in[2];
  const float* Wq     = (const float*)d_in[3];
  const float* Wk     = (const float*)d_in[4];
  const float* Wv     = (const float*)d_in[5];
  const float* Wo     = (const float*)d_in[6];
  float* out = (float*)d_out;

  char* w = (char*)d_ws;
  size_t off = 0;
  auto carve = [&](size_t bytes) -> char* {
    char* p = w + off;
    off = (off + bytes + 255) & ~(size_t)255;
    return p;
  };
  _Float16* hs_h = (_Float16*)carve((size_t)S_LEN * HIDN * 2);
  _Float16* Wq_h = (_Float16*)carve((size_t)NHEAD * DHEAD * HIDN * 2);
  _Float16* Wk_h = (_Float16*)carve((size_t)NKV * DHEAD * HIDN * 2);
  _Float16* Wv_h = (_Float16*)carve((size_t)NKV * DHEAD * HIDN * 2);
  _Float16* Wo_h = (_Float16*)carve((size_t)HIDN * NHEAD * DHEAD * 2);
  _Float16* Qb   = (_Float16*)carve((size_t)S_LEN * NHEAD * DHEAD * 2);
  _Float16* Kb   = (_Float16*)carve((size_t)S_LEN * NKV * DHEAD * 2);
  _Float16* Vb   = (_Float16*)carve((size_t)S_LEN * NKV * DHEAD * 2);
  _Float16* At   = (_Float16*)carve((size_t)S_LEN * NHEAD * DHEAD * 2);

  auto cvt = [&](const float* s, _Float16* d, int n) {
    int blocks = (n / 4 + 255) / 256;
    cvt_kernel<<<blocks, 256, 0, stream>>>(s, d, n);
  };
  cvt(hs, hs_h, S_LEN * HIDN);
  cvt(Wq, Wq_h, NHEAD * DHEAD * HIDN);
  cvt(Wk, Wk_h, NKV * DHEAD * HIDN);
  cvt(Wv, Wv_h, NKV * DHEAD * HIDN);
  cvt(Wo, Wo_h, HIDN * NHEAD * DHEAD);

  gemm_nt<false><<<dim3(HIDN / 128, S_LEN / 128), 256, 0, stream>>>(
      hs_h, Wq_h, Qb, S_LEN, NHEAD * DHEAD, HIDN, HIDN, HIDN, NHEAD * DHEAD);
  gemm_nt<false><<<dim3((NKV * DHEAD) / 128, S_LEN / 128), 256, 0, stream>>>(
      hs_h, Wk_h, Kb, S_LEN, NKV * DHEAD, HIDN, HIDN, HIDN, NKV * DHEAD);
  gemm_nt<false><<<dim3((NKV * DHEAD) / 128, S_LEN / 128), 256, 0, stream>>>(
      hs_h, Wv_h, Vb, S_LEN, NKV * DHEAD, HIDN, HIDN, HIDN, NKV * DHEAD);

  {
    int total = S_LEN * (NHEAD + NKV) * (DHEAD / 2);
    rope_kernel<<<(total + 255) / 256, 256, 0, stream>>>(Qb, Kb, powers, posids);
  }

  attn_fused<<<dim3(S_LEN / 128, NHEAD), 256, 0, stream>>>(Qb, Kb, Vb, At);

  gemm_nt<true><<<dim3(HIDN / 128, S_LEN / 128), 256, 0, stream>>>(
      At, Wo_h, out, S_LEN, HIDN, NHEAD * DHEAD, NHEAD * DHEAD, NHEAD * DHEAD, HIDN);

  (void)in_sizes; (void)n_in; (void)out_size; (void)ws_size;
}
